// MambaModule_34196529610781
// MI455X (gfx1250) — matrix operand order repaired and checked by reading, not compiled, not run on hardware
//
#include <hip/hip_runtime.h>

// ---------------------------------------------------------------- constants
#define D_MODEL   2048
#define ED        4096
#define D_STATE   16
#define D_CONV    4
#define DT_RANK   128
#define N_HEADS   4
#define HEAD_DIM  512       // D_MODEL / N_HEADS
#define BB        4
#define LL        256
#define MTOK      (BB*LL)   // 1024 tokens
#define N_LAYERS  2

// ---------------------------------------------------------------- types
typedef __attribute__((ext_vector_type(16))) __bf16        v16bf;
typedef __attribute__((ext_vector_type(8)))  float         v8f;
typedef __attribute__((ext_vector_type(4)))  unsigned int  u32x4;

union FragU { u32x4 u[2]; v16bf v; };

__device__ __forceinline__ unsigned short f32_to_bf16(float f) {
    unsigned u = __float_as_uint(f);
    unsigned r = u + 0x7FFFu + ((u >> 16) & 1u);   // round-to-nearest-even
    return (unsigned short)(r >> 16);
}

// ---------------------------------------------------------------- generic batched WMMA GEMM
// C[m,n] = alpha * sum_k A[m,k] * W[n,k] + bias[n]
// A: bf16 M x K (lda), W: bf16 N x K (ldb), C: f32 (ldc).
// Block tile: 256(M) x 64(N); 8 waves, each wave owns 32(M) x 64(N) = 2x4 WMMA tiles,
// so every B fragment is reused by two WMMAs (1.5 b128 loads per WMMA).
// Per-z base offsets: base + (z/bdiv)*s?0 + (z%bdiv)*s?1
__global__ __launch_bounds__(256) void gemm_bf16_wmma(
    const unsigned short* __restrict__ A, const unsigned short* __restrict__ Bw,
    float* __restrict__ C, const float* __restrict__ bias,
    int M, int N, int K, int lda, int ldb, int ldc,
    long long sA0, long long sA1, long long sB0, long long sB1,
    long long sC0, long long sC1, int bdiv, float alpha)
{
    const int z = blockIdx.z;
    const unsigned short* Ab = A  + (long long)(z / bdiv) * sA0 + (long long)(z % bdiv) * sA1;
    const unsigned short* Bb = Bw + (long long)(z / bdiv) * sB0 + (long long)(z % bdiv) * sB1;
    float*                Cb = C  + (long long)(z / bdiv) * sC0 + (long long)(z % bdiv) * sC1;

    const int wave = threadIdx.x >> 5;   // 0..7
    const int lane = threadIdx.x & 31;
    const int l16  = lane & 15;
    const int lh   = lane >> 4;          // lane half (K-split selector)

    const int mBase = blockIdx.y * 256 + wave * 32;
    const int nBase = blockIdx.x * 64;
    if (mBase >= M) return;              // wave-uniform; EXEC stays all-ones for WMMA

    const unsigned short* arow0 = Ab + (long long)(mBase + l16) * lda;
    const unsigned short* arow1 = arow0 + (long long)16 * lda;

    int nIdx[4];
    const unsigned short* brow[4];
#pragma unroll
    for (int j = 0; j < 4; ++j) {
        nIdx[j] = nBase + j * 16 + l16;
        int nc  = (nIdx[j] < N) ? nIdx[j] : (N - 1);   // clamp: garbage cols never stored
        brow[j] = Bb + (long long)nc * ldb;
    }

    const v8f zeroV = {0.f,0.f,0.f,0.f,0.f,0.f,0.f,0.f};
    v8f acc0[4] = {zeroV, zeroV, zeroV, zeroV};   // rows mBase..mBase+15
    v8f acc1[4] = {zeroV, zeroV, zeroV, zeroV};   // rows mBase+16..mBase+31

    for (int k = 0; k < K; k += 32) {
        // speculative prefetch two K-steps ahead (global_prefetch_b8; dropped if OOB)
        __builtin_prefetch((const void*)(arow0 + k + 64), 0, 1);
        __builtin_prefetch((const void*)(arow1 + k + 64), 0, 1);

        // A fragments (16x32 bf16): lower lanes hold K {0..7,16..23}, upper {8..15,24..31}
        FragU fa0, fa1;
        fa0.u[0] = *(const u32x4*)(arow0 + k + lh * 8);
        fa0.u[1] = *(const u32x4*)(arow0 + k + 16 + lh * 8);
        fa1.u[0] = *(const u32x4*)(arow1 + k + lh * 8);
        fa1.u[1] = *(const u32x4*)(arow1 + k + 16 + lh * 8);
#pragma unroll
        for (int j = 0; j < 4; ++j) {
            // B fragment (32x16 bf16): lower lanes hold K {0..7,16..23}, upper {8..15,24..31}
            FragU fb;
            fb.u[0] = *(const u32x4*)(brow[j] + k + lh * 8);
            fb.u[1] = *(const u32x4*)(brow[j] + k + 16 + lh * 8);
            acc0[j] = __builtin_amdgcn_wmma_f32_16x16x32_bf16(
                          false, fa0.v, false, fb.v, (short)0, acc0[j], false, false);
            acc1[j] = __builtin_amdgcn_wmma_f32_16x16x32_bf16(
                          false, fa1.v, false, fb.v, (short)0, acc1[j], false, false);
        }
    }

#pragma unroll
    for (int j = 0; j < 4; ++j) {
        if (nIdx[j] >= N) continue;
        float bv = bias ? bias[nIdx[j]] : 0.f;
#pragma unroll
        for (int r = 0; r < 8; ++r) {
            int m0 = mBase + r + lh * 8;          // D layout: vgpr r -> M=r (lo) / r+8 (hi)
            Cb[(long long)m0 * ldc + nIdx[j]]        = alpha * acc0[j][r] + bv;
            Cb[(long long)(m0 + 16) * ldc + nIdx[j]] = alpha * acc1[j][r] + bv;
        }
    }
}

// ---------------------------------------------------------------- elementwise / small kernels
__global__ void f32_to_bf16_kern(const float* __restrict__ s,
                                 unsigned short* __restrict__ d, long long n) {
    long long i = (long long)blockIdx.x * blockDim.x + threadIdx.x;
    if (i < n) d[i] = f32_to_bf16(s[i]);
}

__global__ void slice_bf16_kern(const float* __restrict__ s, unsigned short* __restrict__ d,
                                int rows, int srcld, int cols) {
    long long i = (long long)blockIdx.x * blockDim.x + threadIdx.x;
    long long total = (long long)rows * cols;
    if (i >= total) return;
    int r = (int)(i / cols), c = (int)(i % cols);
    d[i] = f32_to_bf16(s[(long long)r * srcld + c]);
}

__global__ void add_inplace_kern(float* __restrict__ a, const float* __restrict__ b, long long n) {
    long long i = (long long)blockIdx.x * blockDim.x + threadIdx.x;
    if (i < n) a[i] += b[i];
}

__global__ __launch_bounds__(256) void rmsnorm_bf16_k(const float* __restrict__ h,
    const float* __restrict__ w, unsigned short* __restrict__ xnbf)
{
    __shared__ float red[256];
    int m = blockIdx.x, t = threadIdx.x;
    const float* row = h + (long long)m * D_MODEL;
    float ss = 0.f;
    for (int i = t; i < D_MODEL; i += 256) { float v = row[i]; ss += v * v; }
    red[t] = ss; __syncthreads();
    for (int s = 128; s > 0; s >>= 1) { if (t < s) red[t] += red[t + s]; __syncthreads(); }
    float scale = rsqrtf(red[0] / (float)D_MODEL + 1e-5f);
    for (int i = t; i < D_MODEL; i += 256)
        xnbf[(long long)m * D_MODEL + i] = f32_to_bf16(row[i] * scale * w[i]);
}

// depthwise causal conv (D_CONV=4) + bias + SiLU; xin = first ED cols of xz rows (2*ED wide)
__global__ void conv_silu_k(const float* __restrict__ xz, const float* __restrict__ cw,
                            const float* __restrict__ cb, float* __restrict__ xc,
                            unsigned short* __restrict__ xcbf)
{
    long long idx = (long long)blockIdx.x * blockDim.x + threadIdx.x;
    long long total = (long long)MTOK * ED;
    if (idx >= total) return;
    int e = (int)(idx % ED);
    int m = (int)(idx / ED);
    int b = m / LL, l = m % LL;
    float acc = cb[e];
#pragma unroll
    for (int j = 0; j < D_CONV; ++j) {
        int ls = l - (D_CONV - 1) + j;
        if (ls >= 0)
            acc += cw[e * D_CONV + j] * xz[(long long)(b * LL + ls) * (2 * ED) + e];
    }
    float s = acc / (1.f + __expf(-acc));
    xc[idx] = s;
    xcbf[idx] = f32_to_bf16(s);
}

// selective scan: one thread per (b, e); h[16] kept in registers across L steps
__global__ __launch_bounds__(256) void scan_k(
    const float* __restrict__ dpre, const float* __restrict__ dt_b,
    const float* __restrict__ A_log, const float* __restrict__ dbc,
    const float* __restrict__ xc,   const float* __restrict__ xz,
    const float* __restrict__ Dp,   unsigned short* __restrict__ ybf)
{
    int t = blockIdx.x * blockDim.x + threadIdx.x;
    if (t >= BB * ED) return;
    int b = t / ED, e = t % ED;
    float An[D_STATE], h[D_STATE];
#pragma unroll
    for (int n = 0; n < D_STATE; ++n) { An[n] = -__expf(A_log[e * D_STATE + n]); h[n] = 0.f; }
    float dtb = dt_b[e], dpar = Dp[e];
    for (int l = 0; l < LL; ++l) {
        int m = b * LL + l;
        float x0    = dpre[(long long)m * ED + e] + dtb;
        float delta = (x0 > 20.f) ? x0 : __logf(1.f + __expf(x0));
        float xcv   = xc[(long long)m * ED + e];
        float zv    = xz[(long long)m * (2 * ED) + ED + e];
        const float* Bc = dbc + (long long)m * (DT_RANK + 2 * D_STATE) + DT_RANK;
        const float* Cc = Bc + D_STATE;
        float dx = delta * xcv, y = 0.f;
#pragma unroll
        for (int n = 0; n < D_STATE; ++n) {
            float a = __expf(delta * An[n]);
            h[n] = a * h[n] + dx * Bc[n];
            y += h[n] * Cc[n];
        }
        y += xcv * dpar;
        float sz = zv / (1.f + __expf(-zv));
        ybf[(long long)m * ED + e] = f32_to_bf16(y * sz);
    }
}

__global__ void vtrans_k(const float* __restrict__ qkv, unsigned short* __restrict__ vt)
{
    long long idx = (long long)blockIdx.x * blockDim.x + threadIdx.x;
    long long total = (long long)BB * N_HEADS * HEAD_DIM * LL;
    if (idx >= total) return;
    int k = (int)(idx % LL);
    int n = (int)((idx / LL) % HEAD_DIM);
    int hh = (int)((idx / ((long long)LL * HEAD_DIM)) % N_HEADS);
    int b = (int)(idx / ((long long)LL * HEAD_DIM * N_HEADS));
    vt[idx] = f32_to_bf16(
        qkv[(long long)(b * LL + k) * (3 * D_MODEL) + 2 * D_MODEL + hh * HEAD_DIM + n]);
}

__global__ __launch_bounds__(256) void softmax_bf16_k(const float* __restrict__ s,
                                                      unsigned short* __restrict__ p)
{
    __shared__ float red[256];
    int row = blockIdx.x, t = threadIdx.x;
    float v = s[(long long)row * LL + t];
    red[t] = v; __syncthreads();
    for (int st = 128; st > 0; st >>= 1) { if (t < st) red[t] = fmaxf(red[t], red[t + st]); __syncthreads(); }
    float mx = red[0]; __syncthreads();
    float e = __expf(v - mx);
    red[t] = e; __syncthreads();
    for (int st = 128; st > 0; st >>= 1) { if (t < st) red[t] += red[t + st]; __syncthreads(); }
    float inv = 1.f / red[0];
    p[(long long)row * LL + t] = f32_to_bf16(e * inv);
}

// final head: linear -> layernorm -> elu -> linear -> l2norm (4 rows only)
__global__ __launch_bounds__(1024) void head_k(const float* __restrict__ o,
    const float* __restrict__ w1, const float* __restrict__ b1,
    const float* __restrict__ g,  const float* __restrict__ be,
    const float* __restrict__ w2, const float* __restrict__ b2,
    float* __restrict__ out)
{
    __shared__ float hm[1024];
    __shared__ float red[1024];
    __shared__ float ov[64];
    __shared__ float nrm;
    int b = blockIdx.x, j = threadIdx.x;
    const float* last = o + (long long)(b * LL + LL - 1) * D_MODEL;
    const float* wr = w1 + (long long)j * D_MODEL;
    float acc = b1[j];
    for (int d = 0; d < D_MODEL; ++d) acc += last[d] * wr[d];
    red[j] = acc; __syncthreads();
    for (int s = 512; s > 0; s >>= 1) { if (j < s) red[j] += red[j + s]; __syncthreads(); }
    float mu = red[0] / 1024.f; __syncthreads();
    float d0 = acc - mu;
    red[j] = d0 * d0; __syncthreads();
    for (int s = 512; s > 0; s >>= 1) { if (j < s) red[j] += red[j + s]; __syncthreads(); }
    float var = red[0] / 1024.f;
    float v = d0 * rsqrtf(var + 1e-5f) * g[j] + be[j];
    v = (v > 0.f) ? v : (__expf(v) - 1.f);            // ELU
    hm[j] = v; __syncthreads();
    if (j < 64) {
        float oo = b2[j];
        const float* w2r = w2 + j * 1024;
        for (int d = 0; d < 1024; ++d) oo += hm[d] * w2r[d];
        ov[j] = oo;
    }
    __syncthreads();
    if (j == 0) {
        float ssum = 0.f;
        for (int d = 0; d < 64; ++d) ssum += ov[d] * ov[d];
        nrm = fmaxf(sqrtf(ssum), 1e-12f);
    }
    __syncthreads();
    if (j < 64) out[b * 64 + j] = ov[j] / nrm;
}

// ---------------------------------------------------------------- host orchestration
extern "C" void kernel_launch(void* const* d_in, const int* in_sizes, int n_in,
                              void* d_out, int out_size, void* d_ws, size_t ws_size,
                              hipStream_t stream) {
    (void)in_sizes; (void)n_in; (void)out_size; (void)ws_size;
    const float* x          = (const float*)d_in[0];
    const float* norm_w     = (const float*)d_in[1];
    const float* in_proj_w  = (const float*)d_in[2];
    const float* conv_w     = (const float*)d_in[3];
    const float* conv_b     = (const float*)d_in[4];
    const float* x_proj_w   = (const float*)d_in[5];
    const float* dt_proj_w  = (const float*)d_in[6];
    const float* dt_proj_b  = (const float*)d_in[7];
    const float* A_log      = (const float*)d_in[8];
    const float* D_param    = (const float*)d_in[9];
    const float* out_proj_w = (const float*)d_in[10];
    const float* attn_in_w  = (const float*)d_in[11];
    const float* attn_in_b  = (const float*)d_in[12];
    const float* attn_out_w = (const float*)d_in[13];
    const float* attn_out_b = (const float*)d_in[14];
    const float* w1         = (const float*)d_in[15];
    const float* b1         = (const float*)d_in[16];
    const float* ln_g       = (const float*)d_in[17];
    const float* ln_b       = (const float*)d_in[18];
    const float* w2         = (const float*)d_in[19];
    const float* b2         = (const float*)d_in[20];

    // bump allocator over workspace
    char* wp = (char*)d_ws;
    auto alloc = [&](long long bytes) -> void* {
        void* r = (void*)wp;
        wp += (bytes + 255) & ~255LL;
        return r;
    };
    unsigned short* wb_in  = (unsigned short*)alloc(2LL * 2 * ED * D_MODEL * 2);       // (2,8192,2048) bf16
    unsigned short* wb_xp  = (unsigned short*)alloc(2LL * (DT_RANK + 2 * D_STATE) * ED * 2);
    unsigned short* wb_dt  = (unsigned short*)alloc(2LL * ED * DT_RANK * 2);
    unsigned short* wb_out = (unsigned short*)alloc(2LL * D_MODEL * ED * 2);
    unsigned short* wb_ai  = (unsigned short*)alloc(3LL * D_MODEL * D_MODEL * 2);
    unsigned short* wb_ao  = (unsigned short*)alloc((long long)D_MODEL * D_MODEL * 2);
    float*          h_res  = (float*)alloc((long long)MTOK * D_MODEL * 4);
    unsigned short* xn_bf  = (unsigned short*)alloc((long long)MTOK * D_MODEL * 2);
    float*          xz     = (float*)alloc((long long)MTOK * 2 * ED * 4);   // reused as qkv f32
    float*          xc     = (float*)alloc((long long)MTOK * ED * 4);       // reused as scores
    unsigned short* xc_bf  = (unsigned short*)alloc((long long)MTOK * ED * 2);
    float*          dbc    = (float*)alloc((long long)MTOK * (DT_RANK + 2 * D_STATE) * 4);
    unsigned short* dt_bf  = (unsigned short*)alloc((long long)MTOK * DT_RANK * 2);
    float*          dpre   = (float*)alloc((long long)MTOK * ED * 4);       // reused as ao f32
    unsigned short* y_bf   = (unsigned short*)alloc((long long)MTOK * ED * 2); // reused as ao bf16
    float*          pout   = (float*)alloc((long long)MTOK * D_MODEL * 4);
    unsigned short* qkv_bf = (unsigned short*)alloc((long long)MTOK * 3 * D_MODEL * 2);
    unsigned short* vt_bf  = (unsigned short*)alloc((long long)BB * N_HEADS * HEAD_DIM * LL * 2);
    unsigned short* at_bf  = (unsigned short*)alloc((long long)BB * N_HEADS * LL * LL * 2);

    auto cvt = [&](const float* s, unsigned short* d, long long n) {
        int blocks = (int)((n + 255) / 256);
        f32_to_bf16_kern<<<blocks, 256, 0, stream>>>(s, d, n);
    };
    auto gemm = [&](const unsigned short* A, const unsigned short* W, float* Cp,
                    const float* bias, int M, int N, int K, int lda, int ldb, int ldc,
                    long long sA0, long long sA1, long long sB0, long long sB1,
                    long long sC0, long long sC1, int bdiv, int batches, float alpha) {
        dim3 grid((N + 63) / 64, (M + 255) / 256, batches);
        gemm_bf16_wmma<<<grid, 256, 0, stream>>>(A, W, Cp, bias, M, N, K, lda, ldb, ldc,
                                                 sA0, sA1, sB0, sB1, sC0, sC1, bdiv, alpha);
    };

    // weight downcasts (deterministic; ~270 MB of reads, ~12 us at HBM peak)
    cvt(in_proj_w,  wb_in,  2LL * 2 * ED * D_MODEL);
    cvt(x_proj_w,   wb_xp,  2LL * (DT_RANK + 2 * D_STATE) * ED);
    cvt(dt_proj_w,  wb_dt,  2LL * ED * DT_RANK);
    cvt(out_proj_w, wb_out, 2LL * D_MODEL * ED);
    cvt(attn_in_w,  wb_ai,  3LL * D_MODEL * D_MODEL);
    cvt(attn_out_w, wb_ao,  (long long)D_MODEL * D_MODEL);

    hipMemcpyAsync(h_res, x, (long long)MTOK * D_MODEL * 4, hipMemcpyDeviceToDevice, stream);

    const long long eltMD = (long long)MTOK * D_MODEL;
    for (int i = 0; i < N_LAYERS; ++i) {
        rmsnorm_bf16_k<<<MTOK, 256, 0, stream>>>(h_res, norm_w + i * D_MODEL, xn_bf);
        // xz = xn @ in_proj_w^T : (1024 x 8192)
        gemm(xn_bf, wb_in + (long long)i * 2 * ED * D_MODEL, xz, nullptr,
             MTOK, 2 * ED, D_MODEL, D_MODEL, D_MODEL, 2 * ED, 0,0,0,0,0,0, 1, 1, 1.f);
        conv_silu_k<<<(int)(((long long)MTOK * ED + 255) / 256), 256, 0, stream>>>(
            xz, conv_w + (long long)i * ED * D_CONV, conv_b + i * ED, xc, xc_bf);
        // dbc = xc @ x_proj_w^T : (1024 x 160)
        gemm(xc_bf, wb_xp + (long long)i * (DT_RANK + 2 * D_STATE) * ED, dbc, nullptr,
             MTOK, DT_RANK + 2 * D_STATE, ED, ED, ED, DT_RANK + 2 * D_STATE,
             0,0,0,0,0,0, 1, 1, 1.f);
        slice_bf16_kern<<<(int)(((long long)MTOK * DT_RANK + 255) / 256), 256, 0, stream>>>(
            dbc, dt_bf, MTOK, DT_RANK + 2 * D_STATE, DT_RANK);
        // dpre = dt @ dt_proj_w^T : (1024 x 4096)
        gemm(dt_bf, wb_dt + (long long)i * ED * DT_RANK, dpre, nullptr,
             MTOK, ED, DT_RANK, DT_RANK, DT_RANK, ED, 0,0,0,0,0,0, 1, 1, 1.f);
        scan_k<<<(BB * ED) / 256, 256, 0, stream>>>(
            dpre, dt_proj_b + i * ED, A_log + (long long)i * ED * D_STATE,
            dbc, xc, xz, D_param + i * ED, y_bf);
        // pout = y @ out_proj_w^T : (1024 x 2048)
        gemm(y_bf, wb_out + (long long)i * D_MODEL * ED, pout, nullptr,
             MTOK, D_MODEL, ED, ED, ED, D_MODEL, 0,0,0,0,0,0, 1, 1, 1.f);
        add_inplace_kern<<<(int)((eltMD + 255) / 256), 256, 0, stream>>>(h_res, pout, eltMD);
    }

    // ---------------- attention ----------------
    cvt(h_res, xn_bf, eltMD);
    // qkv = seq @ attn_in_w^T + b : (1024 x 6144)  (reuse xz as f32 qkv)
    gemm(xn_bf, wb_ai, xz, attn_in_b,
         MTOK, 3 * D_MODEL, D_MODEL, D_MODEL, D_MODEL, 3 * D_MODEL, 0,0,0,0,0,0, 1, 1, 1.f);
    cvt(xz, qkv_bf, (long long)MTOK * 3 * D_MODEL);
    vtrans_k<<<(int)(((long long)BB * N_HEADS * HEAD_DIM * LL + 255) / 256), 256, 0, stream>>>(
        xz, vt_bf);
    // scores[b,h] = Q K^T / sqrt(HD)  (reuse xc as f32 scores)
    gemm(qkv_bf, qkv_bf + D_MODEL, xc, nullptr,
         LL, LL, HEAD_DIM, 3 * D_MODEL, 3 * D_MODEL, LL,
         (long long)LL * 3 * D_MODEL, HEAD_DIM,
         (long long)LL * 3 * D_MODEL, HEAD_DIM,
         4LL * LL * LL, (long long)LL * LL,
         N_HEADS, BB * N_HEADS, 0.04419417382415922f);
    softmax_bf16_k<<<BB * N_HEADS * LL, LL, 0, stream>>>(xc, at_bf);
    // ao[b,q,h,:] = attn @ V  (reuse dpre as f32 ao, row-major (1024 x 2048))
    gemm(at_bf, vt_bf, dpre, nullptr,
         LL, HEAD_DIM, LL, LL, LL, D_MODEL,
         4LL * LL * LL, (long long)LL * LL,
         4LL * HEAD_DIM * LL, (long long)HEAD_DIM * LL,
         (long long)LL * D_MODEL, HEAD_DIM,
         N_HEADS, BB * N_HEADS, 1.f);
    cvt(dpre, y_bf, eltMD);
    gemm(y_bf, wb_ao, pout, attn_out_b,
         MTOK, D_MODEL, D_MODEL, D_MODEL, D_MODEL, D_MODEL, 0,0,0,0,0,0, 1, 1, 1.f);
    add_inplace_kern<<<(int)((eltMD + 255) / 256), 256, 0, stream>>>(h_res, pout, eltMD);

    // ---------------- final head ----------------
    head_k<<<BB, 1024, 0, stream>>>(h_res, w1, b1, ln_g, ln_b, w2, b2, (float*)d_out);
}
